// Qwen3MoeDecoderLayer_32933809226596
// MI455X (gfx1250) — compile-verified, matrix operand order checked by reading, not run on hardware
//
#include <hip/hip_runtime.h>

// ---------------------------------------------------------------------------
// Qwen3-MoE decoder layer for gfx1250 (MI455X), bf16 WMMA everywhere.
// B=1, S=1024, D=2048, H=32, KV=4, HD=128, E=8, I=1408, TOP_K=2.
// Memory-bound layer (~360MB streamed @23.3TB/s): weights converted f32->bf16
// in flight, all matmuls on v_wmma_f32_16x16x32_bf16, LDS tiles stored in
// WMMA fragment layout (fragment reads = 2x ds_load_b128), GEMM A tiles moved
// by the Tensor Data Mover (TENSORcnt), attention K tiles by per-lane async
// Global->LDS DMA (ASYNCcnt). GEMM is templated on N so strided staging loads
// use immediate offsets (no per-element address adds / xcnt waits).
// ---------------------------------------------------------------------------

typedef __attribute__((ext_vector_type(16))) __bf16 bf16x16;
typedef __attribute__((ext_vector_type(8)))  float  f32x8;
typedef __attribute__((ext_vector_type(4)))  unsigned int u32x4;
typedef __attribute__((ext_vector_type(8)))  int  i32x8;
typedef __attribute__((ext_vector_type(4)))  int  i32x4;

struct U16x16 { unsigned short s[16]; };

__device__ __forceinline__ unsigned short f2bf(float f) {
    unsigned u = __builtin_bit_cast(unsigned, f);
    unsigned r = u + 0x7FFFu + ((u >> 16) & 1u);   // round-to-nearest-even
    return (unsigned short)(r >> 16);
}

// 32B fragment load (LDS or global), emitted as b128 pairs.
__device__ __forceinline__ bf16x16 ld_frag(const unsigned short* p) {
    const unsigned short* pa = (const unsigned short*)__builtin_assume_aligned(p, 32);
    bf16x16 v;
    __builtin_memcpy(&v, pa, 32);
    return v;
}
__device__ __forceinline__ void st_frag(unsigned short* p, const U16x16& v) {
    unsigned short* pa = (unsigned short*)__builtin_assume_aligned(p, 32);
    __builtin_memcpy(pa, &v, 32);
}

// gfx1250 async Global->LDS DMA (ASYNCcnt), 16 bytes per lane.
__device__ __forceinline__ void async_copy_b128(unsigned short* lds_dst,
                                                const unsigned short* gsrc) {
    unsigned loff = (unsigned)(unsigned long long)(void*)lds_dst;
    asm volatile("global_load_async_to_lds_b128 %0, %1, off"
                 :: "v"(loff), "v"(gsrc)
                 : "memory");
}
__device__ __forceinline__ void wait_async0() {
    asm volatile("s_wait_asynccnt 0" ::: "memory");
}

// gfx1250 Tensor Data Mover: 2D tile of 16-bit elements, Global -> LDS.
// D# fields packed per CDNA5 ISA ch.8 (group0: count/lds/global/type,
// group1: data_size, tensor dims, tile dims, dim0 stride). Issued per-wave.
// This toolchain exposes the 6-arg builtin (g0, g1, g2, g3, g4, cpol);
// groups 2-4 are zero for a 2D tensor (VADDR2/3 NULL case).
__device__ __forceinline__ void tdm_load_2d_b16(unsigned lds_off, const void* gaddr,
                                                unsigned tile_d0, unsigned tile_d1,
                                                unsigned long long stride_elems,
                                                unsigned tensor_d0, unsigned tensor_d1) {
    const unsigned long long ga = (unsigned long long)gaddr;
    u32x4 g0;
    g0[0] = 1u;                                           // count=1, user D#
    g0[1] = lds_off;                                      // lds_addr (bytes)
    g0[2] = (unsigned)(ga & 0xFFFFFFFFu);                 // global_addr[31:0]
    g0[3] = (unsigned)((ga >> 32) & 0x01FFFFFFu) | (2u << 30);  // addr[56:32] | type=2
    i32x8 g1;
    g1[0] = (int)(1u << 16);                              // data_size=1 (2 bytes)
    g1[1] = (int)((tensor_d0 & 0xFFFFu) << 16);           // tensor_dim0[15:0]
    g1[2] = (int)(((tensor_d0 >> 16) & 0xFFFFu) | ((tensor_d1 & 0xFFFFu) << 16));
    g1[3] = (int)(((tensor_d1 >> 16) & 0xFFFFu) | ((tile_d0 & 0xFFFFu) << 16));
    g1[4] = (int)(tile_d1 & 0xFFFFu);                     // tile_dim1 (tile_dim2=0)
    g1[5] = (int)(unsigned)(stride_elems & 0xFFFFFFFFull);
    g1[6] = (int)(unsigned)((stride_elems >> 32) & 0xFFFFull);
    g1[7] = 0;
    const i32x4 z4 = {0, 0, 0, 0};
    const i32x8 z8 = {0, 0, 0, 0, 0, 0, 0, 0};
    __builtin_amdgcn_tensor_load_to_lds(g0, g1, z4, z4, z8, 0);
}
__device__ __forceinline__ void wait_tensor0() {
    __builtin_amdgcn_s_wait_tensorcnt(0);
}

// ---------------------------------------------------------------------------
// RMSNorm over D: out_bf (bf16) and optional out_f (f32). One block per row.
// ---------------------------------------------------------------------------
__global__ void k_rmsnorm(const float* __restrict__ x, const float* __restrict__ w,
                          unsigned short* __restrict__ out_bf, float* __restrict__ out_f,
                          int D) {
    __shared__ float red[256];
    const int row = blockIdx.x;
    const float* xr = x + (size_t)row * D;
    float s = 0.f;
    for (int d = threadIdx.x; d < D; d += 256) { float v = xr[d]; s += v * v; }
    red[threadIdx.x] = s; __syncthreads();
    for (int off = 128; off > 0; off >>= 1) {
        if (threadIdx.x < (unsigned)off) red[threadIdx.x] += red[threadIdx.x + off];
        __syncthreads();
    }
    const float rs = rsqrtf(red[0] / (float)D + 1e-6f);
    for (int d = threadIdx.x; d < D; d += 256) {
        float v = xr[d] * rs * w[d];
        out_bf[(size_t)row * D + d] = f2bf(v);
        if (out_f) out_f[(size_t)row * D + d] = v;
    }
}

// ---------------------------------------------------------------------------
// Per-head RMSNorm (over HD=128) + RoPE, f32 in -> bf16 out. grid (S, NH).
// ---------------------------------------------------------------------------
__global__ void k_qknorm_rope(const float* __restrict__ x,      // [S, NH*HD]
                              const float* __restrict__ nw,     // [HD]
                              const float* __restrict__ cosb,   // [S, HD]
                              const float* __restrict__ sinb,   // [S, HD]
                              unsigned short* __restrict__ out, // [S, NH*HD]
                              int NH, int HD) {
    __shared__ float xn[128];
    __shared__ float red[128];
    const int s = blockIdx.x, hh = blockIdx.y, d = threadIdx.x;
    const float* xr = x + ((size_t)s * NH + hh) * HD;
    const float v = xr[d];
    red[d] = v * v; __syncthreads();
    for (int off = 64; off > 0; off >>= 1) {
        if (d < off) red[d] += red[d + off];
        __syncthreads();
    }
    const float rs = rsqrtf(red[0] / (float)HD + 1e-6f);
    const float xv = v * rs * nw[d];
    xn[d] = xv; __syncthreads();
    const float rot = (d < HD / 2) ? -xn[d + HD / 2] : xn[d - HD / 2];
    const float o = xv * cosb[(size_t)s * HD + d] + rot * sinb[(size_t)s * HD + d];
    out[((size_t)s * NH + hh) * HD + d] = f2bf(o);
}

// ---------------------------------------------------------------------------
// Elementwise f32 -> bf16.
// ---------------------------------------------------------------------------
__global__ void k_f32_to_bf16(const float* __restrict__ in, unsigned short* __restrict__ out,
                              int n) {
    int i = blockIdx.x * 256 + threadIdx.x;
    if (i < n) out[i] = f2bf(in[i]);
}

// ---------------------------------------------------------------------------
// bf16 x f32-weight GEMM via v_wmma_f32_16x16x32_bf16, templated on N so the
// strided B staging loads use immediate offsets (j*N*4 folds into IOFFSET).
//   C[M,N](f32) {=, +=} A[M,K](bf16) @ B[K,N](f32->bf16) [+ Cin]
// Block = 128 threads = 4 waves; tile 64x64x32; each wave does a 16x64 stripe.
// A tile moved by the Tensor Data Mover (one wave issues, TENSORcnt); B tile
// staged transposed (BsT[n][k]) so fragment reads are contiguous 32B.
// M%64==0, N%64==0, K%32==0 assumed.
// ---------------------------------------------------------------------------
template <int N>
__global__ void k_gemm_bf16w(const unsigned short* __restrict__ A,
                             const float* __restrict__ B,
                             float* __restrict__ C,
                             const float* __restrict__ Cin,
                             int M, int K, int accumulate) {
    __shared__ __align__(64) unsigned short As[64][32];   // row-major A tile
    __shared__ __align__(64) unsigned short BsT[64][32];  // BsT[n][k]
    const int tid  = threadIdx.x;
    const int wave = tid >> 5;
    const int lane = tid & 31;
    const int lr   = lane & 15;     // fragment row/col index
    const int lh   = lane >> 4;     // K-half selector
    const int m0   = blockIdx.y * 64;
    const int n0   = blockIdx.x * 64;
    const unsigned as_lds = (unsigned)(unsigned long long)(void*)&As[0][0];

    f32x8 acc[4] = {};
    for (int k0 = 0; k0 < K; k0 += 32) {
        if (wave == 0) {   // A tile (64x32 bf16) via TDM: one descriptor, one DMA
            tdm_load_2d_b16(as_lds, A + (size_t)m0 * K + k0,
                            /*tile_d0=*/32, /*tile_d1=*/64,
                            /*stride=*/(unsigned long long)K,
                            /*tensor_d0=*/(unsigned)K, /*tensor_d1=*/(unsigned)M);
        }
        {   // B tile transposed + converted: thread owns column (n0+c), k-range kh*16
            const int c = tid & 63, kh = tid >> 6;
            const float* bp = B + (size_t)(k0 + kh * 16) * N + n0 + c;
            U16x16 t;
            #pragma unroll
            for (int j = 0; j < 16; ++j)     // immediate offsets: one addr reg
                t.s[j] = f2bf(bp[(size_t)j * N]);
            st_frag(&BsT[c][kh * 16], t);
        }
        if (k0 + 32 < K)   // gfx1250 global_prefetch_b8 for next B tile
            __builtin_prefetch(B + (size_t)(k0 + 32 + (tid >> 2)) * N + n0, 0, 1);
        if (wave == 0) wait_tensor0();
        __syncthreads();

        // A fragment: row (wave*16+lr), K = lh*16 + j  -> contiguous 32B
        const bf16x16 a = ld_frag(&As[wave * 16 + lr][lh * 16]);
        #pragma unroll
        for (int nt = 0; nt < 4; ++nt) {
            // B fragment: column (nt*16+lr), K = lh*16 + j -> contiguous 32B
            const bf16x16 b = ld_frag(&BsT[nt * 16 + lr][lh * 16]);
            acc[nt] = __builtin_amdgcn_wmma_f32_16x16x32_bf16(
                false, a, false, b, (short)0, acc[nt], false, false);
        }
        __syncthreads();
    }

    // C layout: VGPR r -> row (r + 8*lh), col = lr within 16-wide subtile
    #pragma unroll
    for (int nt = 0; nt < 4; ++nt) {
        #pragma unroll
        for (int r = 0; r < 8; ++r) {
            const int gr = m0 + wave * 16 + r + 8 * lh;
            const int gc = n0 + nt * 16 + lr;
            const size_t idx = (size_t)gr * N + gc;
            float v = acc[nt][r];
            if (Cin) v += Cin[idx];
            if (accumulate) C[idx] += v; else C[idx] = v;
        }
    }
}

// ---------------------------------------------------------------------------
// Flash-attention (causal, GQA), templated on H/KV/HD so all strides are
// compile-time. grid (S/64, H), block 128 (4 waves). Each wave owns 16 query
// rows. QK^T and P*V via bf16 WMMA; online softmax. K tiles staged by per-lane
// async DMA (row-major = B-fragment layout for QK^T); V staged transposed
// (VsT[d][key]) = B-fragment layout for P*V. HD must be 128.
// ---------------------------------------------------------------------------
template <int H, int KV, int HD>
__global__ void k_attention(const unsigned short* __restrict__ qb, // [S, H*HD]
                            const unsigned short* __restrict__ kb, // [S, KV*HD]
                            const unsigned short* __restrict__ vb, // [S, KV*HD]
                            unsigned short* __restrict__ ob,       // [S, H*HD]
                            int S, float scale) {
    __shared__ __align__(64) unsigned short Ks[64][HD];    // [key][d]
    __shared__ __align__(64) unsigned short VsT[HD][64];   // [d][key]
    __shared__ __align__(64) unsigned short Ps[4][16][64]; // per-wave P tile

    constexpr int qstride = H * HD;
    constexpr int kstride = KV * HD;
    const int tid  = threadIdx.x;
    const int wave = tid >> 5;
    const int lane = tid & 31;
    const int lr   = lane & 15;
    const int lh   = lane >> 4;
    const int qt   = blockIdx.x;
    const int h    = blockIdx.y;
    const int kvh  = h / (H / KV);
    const int q0   = qt * 64;

    // Q fragments resident in VGPRs: row q0+wave*16+lr, d = ks*32 + lh*16 + j
    bf16x16 aq[4];
    {
        const unsigned short* qrow =
            qb + (size_t)(q0 + wave * 16 + lr) * qstride + (size_t)h * HD;
        #pragma unroll
        for (int ks = 0; ks < 4; ++ks)
            aq[ks] = ld_frag(qrow + ks * 32 + lh * 16);
    }

    f32x8 o[8] = {};
    float m_[8], l_[8];
    #pragma unroll
    for (int r = 0; r < 8; ++r) { m_[r] = -1e30f; l_[r] = 0.f; }

    for (int kt = 0; kt <= qt; ++kt) {
        const int k0 = kt * 64;
        {   // K tile: pure copy via async DMA; thread -> 128B of one key row
            const int r  = tid >> 1;
            const int c0 = (tid & 1) * 64;
            const unsigned short* ksrc =
                kb + (size_t)(k0 + r) * kstride + (size_t)kvh * HD + c0;
            #pragma unroll
            for (int i = 0; i < 64; i += 8)
                async_copy_b128(&Ks[r][c0 + i], ksrc + i);
        }
        {   // V tile transposed through registers: thread owns head-dim d = tid
            const int d = tid;
            const unsigned short* vcol =
                vb + (size_t)k0 * kstride + (size_t)kvh * HD + d;
            #pragma unroll
            for (int j0 = 0; j0 < 64; j0 += 16) {
                U16x16 t;
                #pragma unroll
                for (int j = 0; j < 16; ++j)   // immediate offsets (kstride const)
                    t.s[j] = vcol[(size_t)(j0 + j) * kstride];
                st_frag(&VsT[d][j0], t);
            }
        }
        wait_async0();
        __syncthreads();

        // S = Q @ K^T : 16 (q rows) x 64 (keys) per wave
        f32x8 sc[4] = {};
        #pragma unroll
        for (int nt = 0; nt < 4; ++nt) {
            #pragma unroll
            for (int ks = 0; ks < 4; ++ks) {
                // B fragment: key nt*16+lr, d = ks*32+lh*16+j -> contiguous 32B
                const bf16x16 b = ld_frag(&Ks[nt * 16 + lr][ks * 32 + lh * 16]);
                sc[nt] = __builtin_amdgcn_wmma_f32_16x16x32_bf16(
                    false, aq[ks], false, b, (short)0, sc[nt], false, false);
            }
        }

        // causal mask + scale, row-max
        float rm[8];
        #pragma unroll
        for (int r = 0; r < 8; ++r) rm[r] = -1e30f;
        #pragma unroll
        for (int nt = 0; nt < 4; ++nt) {
            #pragma unroll
            for (int r = 0; r < 8; ++r) {
                const int grow = q0 + wave * 16 + r + 8 * lh;
                const int gkey = k0 + nt * 16 + lr;
                float v = sc[nt][r] * scale;
                if (gkey > grow) v = -1e30f;
                sc[nt][r] = v;
                rm[r] = fmaxf(rm[r], v);
            }
        }
        #pragma unroll
        for (int off = 1; off < 16; off <<= 1)
            #pragma unroll
            for (int r = 0; r < 8; ++r)
                rm[r] = fmaxf(rm[r], __shfl_xor(rm[r], off, 32));

        float alpha[8], rs_[8];
        #pragma unroll
        for (int r = 0; r < 8; ++r) {
            const float mn = fmaxf(m_[r], rm[r]);
            alpha[r] = __expf(m_[r] - mn);
            m_[r] = mn;
            rs_[r] = 0.f;
        }
        // exponentiate, row-sum, stash P (bf16) in per-wave LDS
        #pragma unroll
        for (int nt = 0; nt < 4; ++nt) {
            #pragma unroll
            for (int r = 0; r < 8; ++r) {
                const float p = __expf(sc[nt][r] - m_[r]);
                rs_[r] += p;
                Ps[wave][r + 8 * lh][nt * 16 + lr] = f2bf(p);
            }
        }
        #pragma unroll
        for (int off = 1; off < 16; off <<= 1)
            #pragma unroll
            for (int r = 0; r < 8; ++r)
                rs_[r] += __shfl_xor(rs_[r], off, 32);
        #pragma unroll
        for (int r = 0; r < 8; ++r) l_[r] = l_[r] * alpha[r] + rs_[r];
        #pragma unroll
        for (int c = 0; c < 8; ++c)
            #pragma unroll
            for (int r = 0; r < 8; ++r)
                o[c][r] *= alpha[r];

        __syncthreads();   // Ps visible for fragment reads

        // O += P @ V : K-dim = 64 keys (2 WMMA steps), N = 128 head dims
        #pragma unroll
        for (int ks2 = 0; ks2 < 2; ++ks2) {
            // A fragment: row lr, key = ks2*32 + lh*16 + j -> contiguous 32B
            const bf16x16 ap = ld_frag(&Ps[wave][lr][ks2 * 32 + lh * 16]);
            #pragma unroll
            for (int c = 0; c < 8; ++c) {
                // B fragment: d = c*16+lr, key = ks2*32+lh*16+j -> contiguous 32B
                const bf16x16 b = ld_frag(&VsT[c * 16 + lr][ks2 * 32 + lh * 16]);
                o[c] = __builtin_amdgcn_wmma_f32_16x16x32_bf16(
                    false, ap, false, b, (short)0, o[c], false, false);
            }
        }
        __syncthreads();   // before next tile overwrites Ks/VsT
    }

    // normalize + store bf16
    #pragma unroll
    for (int c = 0; c < 8; ++c) {
        #pragma unroll
        for (int r = 0; r < 8; ++r) {
            const int grow = q0 + wave * 16 + r + 8 * lh;
            const float denom = (l_[r] > 0.f) ? l_[r] : 1.f;
            ob[(size_t)grow * qstride + (size_t)h * HD + c * 16 + lr] =
                f2bf(o[c][r] / denom);
        }
    }
}

// ---------------------------------------------------------------------------
// Router: logits = x @ router_w, softmax, top-2, renormalize -> full_w[T,E].
// One block (256 thr) per token.
// ---------------------------------------------------------------------------
__global__ void k_router(const float* __restrict__ x, const float* __restrict__ rw,
                         float* __restrict__ full_w, int D, int E) {
    __shared__ float red[256];
    __shared__ float logits[8];
    const int t = blockIdx.x;
    float acc[8] = {};
    for (int d = threadIdx.x; d < D; d += 256) {
        const float xv = x[(size_t)t * D + d];
        #pragma unroll
        for (int e = 0; e < 8; ++e) acc[e] += xv * rw[(size_t)d * E + e];
    }
    for (int e = 0; e < 8; ++e) {
        red[threadIdx.x] = acc[e]; __syncthreads();
        for (int off = 128; off > 0; off >>= 1) {
            if (threadIdx.x < (unsigned)off) red[threadIdx.x] += red[threadIdx.x + off];
            __syncthreads();
        }
        if (threadIdx.x == 0) logits[e] = red[0];
        __syncthreads();
    }
    if (threadIdx.x == 0) {
        float mx = logits[0];
        for (int e = 1; e < 8; ++e) mx = fmaxf(mx, logits[e]);
        float p[8], sum = 0.f;
        for (int e = 0; e < 8; ++e) { p[e] = __expf(logits[e] - mx); sum += p[e]; }
        int i0 = 0;
        for (int e = 1; e < 8; ++e) if (p[e] > p[i0]) i0 = e;
        int i1 = (i0 == 0) ? 1 : 0;
        for (int e = 0; e < 8; ++e) if (e != i0 && p[e] > p[i1]) i1 = e;
        const float w0 = p[i0] / sum, w1 = p[i1] / sum;
        const float tw = w0 + w1;
        for (int e = 0; e < 8; ++e) full_w[(size_t)t * E + e] = 0.f;
        full_w[(size_t)t * E + i0] = w0 / tw;
        full_w[(size_t)t * E + i1] = w1 / tw;
    }
}

// ---------------------------------------------------------------------------
// act = bf16( silu(g) * u * full_w[t,e] )  — router weight folded in here.
// ---------------------------------------------------------------------------
__global__ void k_silu_mul(const float* __restrict__ g, const float* __restrict__ u,
                           const float* __restrict__ fw, unsigned short* __restrict__ act,
                           int I, int E, int e, int total) {
    const int i = blockIdx.x * 256 + threadIdx.x;
    if (i >= total) return;
    const int t = i / I;
    const float gv = g[i];
    const float s = gv / (1.f + __expf(-gv));
    act[i] = f2bf(s * u[i] * fw[(size_t)t * E + e]);
}

// ---------------------------------------------------------------------------
// Host orchestration
// ---------------------------------------------------------------------------
extern "C" void kernel_launch(void* const* d_in, const int* in_sizes, int n_in,
                              void* d_out, int out_size, void* d_ws, size_t ws_size,
                              hipStream_t stream) {
    (void)in_sizes; (void)n_in; (void)out_size; (void)ws_size;
    const float* hidden   = (const float*)d_in[0];
    const float* w_q      = (const float*)d_in[1];
    const float* w_k      = (const float*)d_in[2];
    const float* w_v      = (const float*)d_in[3];
    const float* w_o      = (const float*)d_in[4];
    const float* qn_w     = (const float*)d_in[5];
    const float* kn_w     = (const float*)d_in[6];
    const float* ln1_w    = (const float*)d_in[7];
    const float* ln2_w    = (const float*)d_in[8];
    const float* router_w = (const float*)d_in[9];
    const float* w_gate   = (const float*)d_in[10];
    const float* w_up     = (const float*)d_in[11];
    const float* w_down   = (const float*)d_in[12];
    const float* cosb     = (const float*)d_in[13];
    const float* sinb     = (const float*)d_in[14];
    float* out = (float*)d_out;

    constexpr int S = 1024, D = 2048, H = 32, KV = 4, HD = 128, E = 8, I = 1408;
    const float scale = 0.08838834764831845f;   // HD^-0.5

    char* ws = (char*)d_ws;
    auto alloc = [&](size_t bytes) -> char* {
        char* p = ws;
        ws += (bytes + 255) & ~(size_t)255;
        return p;
    };
    unsigned short* xn_bf  = (unsigned short*)alloc((size_t)S * D * 2);
    float*          q_f32  = (float*)alloc((size_t)S * H * HD * 4);
    float*          k_f32  = (float*)alloc((size_t)S * KV * HD * 4);
    float*          v_f32  = (float*)alloc((size_t)S * KV * HD * 4);
    unsigned short* q_bf   = (unsigned short*)alloc((size_t)S * H * HD * 2);
    unsigned short* k_bf   = (unsigned short*)alloc((size_t)S * KV * HD * 2);
    unsigned short* v_bf   = (unsigned short*)alloc((size_t)S * KV * HD * 2);
    unsigned short* attn_bf= (unsigned short*)alloc((size_t)S * H * HD * 2);
    float*          h_f32  = (float*)alloc((size_t)S * D * 4);
    unsigned short* x2_bf  = (unsigned short*)alloc((size_t)S * D * 2);
    float*          x2_f32 = (float*)alloc((size_t)S * D * 4);
    float*          full_w = (float*)alloc((size_t)S * E * 4);
    float*          g_f32  = (float*)alloc((size_t)S * I * 4);
    float*          u_f32  = (float*)alloc((size_t)S * I * 4);
    unsigned short* act_bf = (unsigned short*)alloc((size_t)S * I * 2);

    // 1) pre-attn RMSNorm -> bf16
    k_rmsnorm<<<S, 256, 0, stream>>>(hidden, ln1_w, xn_bf, nullptr, D);

    // 2) Q/K/V projections (bf16 WMMA), N templated per shape
    k_gemm_bf16w<H * HD><<<dim3((H * HD) / 64, S / 64), 128, 0, stream>>>(
        xn_bf, w_q, q_f32, nullptr, S, D, 0);
    k_gemm_bf16w<KV * HD><<<dim3((KV * HD) / 64, S / 64), 128, 0, stream>>>(
        xn_bf, w_k, k_f32, nullptr, S, D, 0);
    k_gemm_bf16w<KV * HD><<<dim3((KV * HD) / 64, S / 64), 128, 0, stream>>>(
        xn_bf, w_v, v_f32, nullptr, S, D, 0);

    // 3) per-head norm + RoPE -> bf16; V just converts
    k_qknorm_rope<<<dim3(S, H), HD, 0, stream>>>(q_f32, qn_w, cosb, sinb, q_bf, H, HD);
    k_qknorm_rope<<<dim3(S, KV), HD, 0, stream>>>(k_f32, kn_w, cosb, sinb, k_bf, KV, HD);
    k_f32_to_bf16<<<(S * KV * HD + 255) / 256, 256, 0, stream>>>(v_f32, v_bf, S * KV * HD);

    // 4) causal flash attention (WMMA for QK^T and P*V)
    k_attention<H, KV, HD><<<dim3(S / 64, H), 128, 0, stream>>>(
        q_bf, k_bf, v_bf, attn_bf, S, scale);

    // 5) output projection + residual: h = hidden + attn @ w_o
    k_gemm_bf16w<D><<<dim3(D / 64, S / 64), 128, 0, stream>>>(
        attn_bf, w_o, h_f32, hidden, S, H * HD, 0);

    // 6) post-attn RMSNorm (bf16 for GEMMs, f32 for router)
    k_rmsnorm<<<S, 256, 0, stream>>>(h_f32, ln2_w, x2_bf, x2_f32, D);

    // 7) router -> normalized top-2 weights (full E-width, zeros elsewhere)
    k_router<<<S, 256, 0, stream>>>(x2_f32, router_w, full_w, D, E);

    // 8) seed output with residual h, then accumulate expert contributions
    (void)hipMemcpyAsync(out, h_f32, (size_t)S * D * sizeof(float),
                         hipMemcpyDeviceToDevice, stream);
    for (int e = 0; e < E; ++e) {
        const float* wg = w_gate + (size_t)e * D * I;
        const float* wu = w_up   + (size_t)e * D * I;
        const float* wd = w_down + (size_t)e * I * D;
        k_gemm_bf16w<I><<<dim3(I / 64, S / 64), 128, 0, stream>>>(
            x2_bf, wg, g_f32, nullptr, S, D, 0);
        k_gemm_bf16w<I><<<dim3(I / 64, S / 64), 128, 0, stream>>>(
            x2_bf, wu, u_f32, nullptr, S, D, 0);
        k_silu_mul<<<(S * I + 255) / 256, 256, 0, stream>>>(
            g_f32, u_f32, full_w, act_bf, I, E, e, S * I);
        k_gemm_bf16w<D><<<dim3(D / 64, S / 64), 128, 0, stream>>>(
            act_bf, wd, out, nullptr, S, I, 1);
    }
}
